// RNNLMModel_523986010563
// MI455X (gfx1250) — matrix operand order repaired and checked by reading, not compiled, not run on hardware
//
#include <hip/hip_runtime.h>

// ---------------------------------------------------------------------------
// Types for CDNA5 WMMA (wave32): v_wmma_f32_16x16x32_bf16
// ---------------------------------------------------------------------------
typedef __bf16 bf16;
typedef __attribute__((ext_vector_type(8)))  bf16  v8bf;
typedef __attribute__((ext_vector_type(16))) bf16  v16bf;
typedef __attribute__((ext_vector_type(8)))  float v8f;

__device__ __forceinline__ v8f wmma_bf16(v16bf a, v16bf b, v8f c) {
    // 8 args: (neg_a, A, neg_b, B, c_mod, C, reuse_a, reuse_b)
    return __builtin_amdgcn_wmma_f32_16x16x32_bf16(false, a, false, b,
                                                   (short)0, c, false, false);
}

// A operand (16x32 bf16, M x K): lane l holds row (mbase + (l&15));
// 16 contiguous K values starting at kbase + hi*16, hi=l>>4.
__device__ __forceinline__ v16bf load_a_bf16(const bf16* __restrict__ base,
                                             int ld, int mbase, int kbase, int lane) {
    const int r  = mbase + (lane & 15);
    const int hi = lane >> 4;
    const bf16* p0 = base + (long)r * ld + kbase + hi * 16;
    v8bf lo = *(const v8bf*)p0;          // 16B
    v8bf hg = *(const v8bf*)(p0 + 8);    // 16B
    return __builtin_shufflevector(lo, hg, 0,1,2,3,4,5,6,7,8,9,10,11,12,13,14,15);
}

// B operand (32x16, K x N) with B = W^T: lane l holds W row (nbase + (l&15)),
// 16 contiguous K values starting at kbase + (l>>4)*16. W is f32 in memory;
// convert to bf16 in-register. Row index clamped to rowmax (for V tail tiles).
__device__ __forceinline__ v16bf load_b_f32(const float* __restrict__ base,
                                            int ld, int nbase, int kbase, int lane,
                                            int rowmax) {
    int r = nbase + (lane & 15);
    r = (r < rowmax) ? r : rowmax;
    const int hi = lane >> 4;
    const float* p = base + (long)r * ld + kbase + hi * 16;
    v8f lo = *(const v8f*)p;
    v8f hg = *(const v8f*)(p + 8);
    v16bf b;
#pragma unroll
    for (int i = 0; i < 8; ++i) {
        b[i]     = (bf16)lo[i];
        b[i + 8] = (bf16)hg[i];
    }
    return b;
}

__device__ __forceinline__ float fast_sigmoid(float x) {
    return 1.0f / (1.0f + __expf(-x));
}

// ---------------------------------------------------------------------------
// Prep: embedding gather -> bf16, hidden (4 x B x H) f32 -> bf16
// ---------------------------------------------------------------------------
__global__ void __launch_bounds__(256)
prep_kernel(const int* __restrict__ x, const float* __restrict__ embed,
            const float* __restrict__ hidden,
            bf16* __restrict__ Xbf, bf16* __restrict__ Hbf) {
    const int BE = 1024 * 512;
    long i = (long)blockIdx.x * blockDim.x + threadIdx.x;   // 0 .. 4*BE-1
    if (i < BE) {
        int b = (int)(i >> 9);
        int e = (int)(i & 511);
        Xbf[i] = (bf16)embed[(long)x[b] * 512 + e];
    }
    Hbf[i] = (bf16)hidden[i];
}

// ---------------------------------------------------------------------------
// Fused GRU cell: one wave computes a 32(B) x 16(H) strip (MT=2 M-tiles),
// so each converted gate B-tile feeds 2 WMMAs.
//   gi = x @ Wih^T + bih ; gh = h @ Whh^T + bhh
//   r = sig(gi_r+gh_r); z = sig(gi_z+gh_z); n = tanh(gi_n + r*gh_n)
//   h' = (1-z)*n + z*h
// ---------------------------------------------------------------------------
__global__ void __launch_bounds__(256)
gru_cell_kernel(const bf16* __restrict__ X, int ldx, int K1,
                const bf16* __restrict__ Hb,
                const float* __restrict__ hprev,
                const float* __restrict__ Wih, const float* __restrict__ Whh,
                const float* __restrict__ bih, const float* __restrict__ bhh,
                float* __restrict__ hout, bf16* __restrict__ hbout, int ldo) {
    const int H = 512;
    const int wid  = blockIdx.x * (blockDim.x >> 5) + (threadIdx.x >> 5);
    const int lane = threadIdx.x & 31;
    const int bt = wid >> 5;        // B/32 = 32 strips,  H/16 = 32 tiles per row
    const int nt = wid & 31;
    const int mbase = bt * 32, nbase = nt * 16;

    v8f ar[2] = {}, az[2] = {}, ain[2] = {}, ahn[2] = {};

    // gi = X @ Wih^T  (K = K1): per k-step, 3 B conversions feed 6 WMMAs
    for (int k = 0; k < K1; k += 32) {
        v16bf a0 = load_a_bf16(X, ldx, mbase,      k, lane);
        v16bf a1 = load_a_bf16(X, ldx, mbase + 16, k, lane);
        v16bf br = load_b_f32(Wih,                   K1, nbase, k, lane, 3 * H - 1);
        ar[0]  = wmma_bf16(a0, br, ar[0]);
        ar[1]  = wmma_bf16(a1, br, ar[1]);
        v16bf bz = load_b_f32(Wih + 512 * (long)K1,  K1, nbase, k, lane, 3 * H - 1);
        az[0]  = wmma_bf16(a0, bz, az[0]);
        az[1]  = wmma_bf16(a1, bz, az[1]);
        v16bf bn = load_b_f32(Wih + 1024 * (long)K1, K1, nbase, k, lane, 3 * H - 1);
        ain[0] = wmma_bf16(a0, bn, ain[0]);
        ain[1] = wmma_bf16(a1, bn, ain[1]);
    }
    // gh = H @ Whh^T  (K = H)
    for (int k = 0; k < H; k += 32) {
        v16bf a0 = load_a_bf16(Hb, H, mbase,      k, lane);
        v16bf a1 = load_a_bf16(Hb, H, mbase + 16, k, lane);
        v16bf br = load_b_f32(Whh,            H, nbase, k, lane, 3 * H - 1);
        ar[0]  = wmma_bf16(a0, br, ar[0]);
        ar[1]  = wmma_bf16(a1, br, ar[1]);
        v16bf bz = load_b_f32(Whh + 512 * H,  H, nbase, k, lane, 3 * H - 1);
        az[0]  = wmma_bf16(a0, bz, az[0]);
        az[1]  = wmma_bf16(a1, bz, az[1]);
        v16bf bn = load_b_f32(Whh + 1024 * H, H, nbase, k, lane, 3 * H - 1);
        ahn[0] = wmma_bf16(a0, bn, ahn[0]);
        ahn[1] = wmma_bf16(a1, bn, ahn[1]);
    }

    // Epilogue: C/D layout => n = nbase + (lane&15), m = tile_m + i + 8*(lane>=16)
    const int ncol = nbase + (lane & 15);
    const float brv = bih[ncol]         + bhh[ncol];
    const float bzv = bih[H + ncol]     + bhh[H + ncol];
    const float biv = bih[2 * H + ncol];
    const float bhv = bhh[2 * H + ncol];
#pragma unroll
    for (int t = 0; t < 2; ++t) {
        const int m0 = mbase + t * 16 + ((lane >> 4) << 3);
#pragma unroll
        for (int i = 0; i < 8; ++i) {
            const int m = m0 + i;
            float r  = fast_sigmoid(ar[t][i] + brv);
            float z  = fast_sigmoid(az[t][i] + bzv);
            float hp = hprev[(long)m * H + ncol];
            float nn = tanhf(ain[t][i] + biv + r * (ahn[t][i] + bhv));
            float h  = (1.0f - z) * nn + z * hp;
            hout[(long)m * H + ncol]    = h;
            hbout[(long)m * ldo + ncol] = (bf16)h;
        }
    }
}

// ---------------------------------------------------------------------------
// Projection: logits[b, v] = hb1[b] . Wproj[v] + bproj[v]
// One wave per 64(B) x 32(V) block: 4 M-tiles x 2 N-tiles, 8 independent
// accumulators. Per 32-K step: 4 A-tile + 2 B-tile loads feed 8 WMMAs
// (0.75 tile-loads/WMMA), each converted Wproj tile feeds 4 WMMAs.
// ---------------------------------------------------------------------------
__global__ void __launch_bounds__(256)
proj_kernel(const bf16* __restrict__ A, const float* __restrict__ Wproj,
            const float* __restrict__ bproj, float* __restrict__ logits, int V) {
    const int K   = 512;
    const int VT2 = 1571;  // ceil(ceil(50257/16)/2) N-tile pairs
    const int wid  = blockIdx.x * (blockDim.x >> 5) + (threadIdx.x >> 5);
    const int lane = threadIdx.x & 31;
    const int bt = wid / VT2;          // 1024/64 = 16 M strips
    const int vt = wid % VT2;
    const int mbase = bt * 64, nbase = vt * 32;

    v8f acc[4][2] = {};
#pragma unroll 2
    for (int k = 0; k < K; k += 32) {
        v16bf b0 = load_b_f32(Wproj, K, nbase,      k, lane, V - 1);
        v16bf b1 = load_b_f32(Wproj, K, nbase + 16, k, lane, V - 1);
        v16bf a0 = load_a_bf16(A, K, mbase,      k, lane);
        v16bf a1 = load_a_bf16(A, K, mbase + 16, k, lane);
        v16bf a2 = load_a_bf16(A, K, mbase + 32, k, lane);
        v16bf a3 = load_a_bf16(A, K, mbase + 48, k, lane);
        acc[0][0] = wmma_bf16(a0, b0, acc[0][0]);
        acc[1][0] = wmma_bf16(a1, b0, acc[1][0]);
        acc[2][0] = wmma_bf16(a2, b0, acc[2][0]);
        acc[3][0] = wmma_bf16(a3, b0, acc[3][0]);
        acc[0][1] = wmma_bf16(a0, b1, acc[0][1]);
        acc[1][1] = wmma_bf16(a1, b1, acc[1][1]);
        acc[2][1] = wmma_bf16(a2, b1, acc[2][1]);
        acc[3][1] = wmma_bf16(a3, b1, acc[3][1]);
    }

#pragma unroll
    for (int nt = 0; nt < 2; ++nt) {
        const int ncol = nbase + nt * 16 + (lane & 15);
        if (ncol < V) {
            const float bp = bproj[ncol];
#pragma unroll
            for (int t = 0; t < 4; ++t) {
                const int m0 = mbase + t * 16 + ((lane >> 4) << 3);
#pragma unroll
                for (int i = 0; i < 8; ++i) {
                    const int m = m0 + i;
                    logits[(long)m * V + ncol] = acc[t][nt][i] + bp;
                }
            }
        }
    }
}

// ---------------------------------------------------------------------------
// In-place row softmax over V=50257. One block per row; online (max,sum)
// accumulation per thread, LDS tree combine, then normalize pass.
// ---------------------------------------------------------------------------
__global__ void __launch_bounds__(256)
softmax_kernel(float* __restrict__ logits, int V) {
    float* row = logits + (long)blockIdx.x * V;
    const int tid = threadIdx.x;

    float m = -3.4e38f, s = 0.0f;
    for (int i = tid; i < V; i += 256) {
        float v = row[i];
        if (v > m) { s *= __expf(m - v); m = v; }
        s += __expf(v - m);
    }
    __shared__ float sm[256], ss[256];
    sm[tid] = m; ss[tid] = s;
    __syncthreads();
    for (int o = 128; o > 0; o >>= 1) {
        if (tid < o) {
            float m1 = sm[tid], s1 = ss[tid];
            float m2 = sm[tid + o], s2 = ss[tid + o];
            float mm = fmaxf(m1, m2);
            ss[tid] = s1 * __expf(m1 - mm) + s2 * __expf(m2 - mm);
            sm[tid] = mm;
        }
        __syncthreads();
    }
    const float M = sm[0];
    const float inv = 1.0f / ss[0];
    for (int i = tid; i < V; i += 256)
        row[i] = __expf(row[i] - M) * inv;
}

// ---------------------------------------------------------------------------
// Launch
// ---------------------------------------------------------------------------
extern "C" void kernel_launch(void* const* d_in, const int* in_sizes, int n_in,
                              void* d_out, int out_size, void* d_ws, size_t ws_size,
                              hipStream_t stream) {
    const int B = 1024, V = 50257, E = 512, H = 512;
    (void)in_sizes; (void)n_in; (void)out_size; (void)ws_size;

    const int*   x      = (const int*)d_in[0];
    const float* hidden = (const float*)d_in[1];
    const float* embed  = (const float*)d_in[2];
    const float* Wih0   = (const float*)d_in[3];
    const float* Whh0   = (const float*)d_in[4];
    const float* bih0   = (const float*)d_in[5];
    const float* bhh0   = (const float*)d_in[6];
    const float* Wih1   = (const float*)d_in[7];
    const float* Whh1   = (const float*)d_in[8];
    const float* bih1   = (const float*)d_in[9];
    const float* bhh1   = (const float*)d_in[10];
    const float* Wproj  = (const float*)d_in[11];
    const float* bproj  = (const float*)d_in[12];

    float* probs = (float*)d_out;                 // B * V
    float* newh  = probs + (long)B * V;           // 4 * B * H  [hf0, hb0, hf1, hb1]

    char* ws = (char*)d_ws;
    bf16* Xbf  = (bf16*)(ws);                        // B*E bf16      (1 MB)
    bf16* Hbf  = (bf16*)(ws + (1l << 20));           // 4*B*H bf16    (4 MB)
    bf16* Xcat = (bf16*)(ws + (5l << 20));           // B*(2H) bf16   (2 MB)
    bf16* Hf1  = (bf16*)(ws + (7l << 20));           // B*H bf16      (1 MB)
    bf16* Hb1  = (bf16*)(ws + (8l << 20));           // B*H bf16      (1 MB)

    // 1) gather + f32->bf16 conversions
    prep_kernel<<<(4 * B * H) / 256, 256, 0, stream>>>(x, embed, hidden, Xbf, Hbf);

    // GRU: (B/32) * (H/16) = 1024 waves, 8 waves per block
    const int gruBlocks = (B / 32) * (H / 16) / 8;   // 128

    // 2) layer 0 forward / backward  (K1 = E)
    gru_cell_kernel<<<gruBlocks, 256, 0, stream>>>(
        Xbf, E, E, Hbf, hidden,
        Wih0, Whh0, bih0, bhh0,
        newh, Xcat, 2 * H);
    gru_cell_kernel<<<gruBlocks, 256, 0, stream>>>(
        Xbf, E, E, Hbf + (long)B * H, hidden + (long)B * H,
        Wih0 + 3l * H * E, Whh0 + 3l * H * H, bih0 + 3 * H, bhh0 + 3 * H,
        newh + (long)B * H, Xcat + H, 2 * H);

    // 3) layer 1 forward / backward  (K1 = 2H)
    gru_cell_kernel<<<gruBlocks, 256, 0, stream>>>(
        Xcat, 2 * H, 2 * H, Hbf + 2l * B * H, hidden + 2l * B * H,
        Wih1, Whh1, bih1, bhh1,
        newh + 2l * B * H, Hf1, H);
    gru_cell_kernel<<<gruBlocks, 256, 0, stream>>>(
        Xcat, 2 * H, 2 * H, Hbf + 3l * B * H, hidden + 3l * B * H,
        Wih1 + 3l * H * 2 * H, Whh1 + 3l * H * H, bih1 + 3 * H, bhh1 + 3 * H,
        newh + 3l * B * H, Hb1, H);

    // 4) vocab projection: 16 M-strips * 1571 N-pairs = 25136 waves, 8/block
    const int projBlocks = (16 * 1571) / 8;          // 3142
    proj_kernel<<<projBlocks, 256, 0, stream>>>(Hb1, Wproj, bproj, probs, V);

    // 5) in-place softmax
    softmax_kernel<<<B, 256, 0, stream>>>(probs, V);
}
